// Discriminator_73753178406914
// MI455X (gfx1250) — matrix operand order repaired and checked by reading, not compiled, not run on hardware
//
#include <hip/hip_runtime.h>
#include <hip/hip_bf16.h>
#include <math.h>

typedef __attribute__((ext_vector_type(16))) __bf16 v16bf;
typedef __attribute__((ext_vector_type(8)))  float  v8f;
typedef __attribute__((ext_vector_type(4))) unsigned int u32x4;
typedef __attribute__((ext_vector_type(8))) unsigned int u32x8;

#define HDIM   512
#define KSTEPS 16      // 512 / 32
#define NLAYER 3

// LDS layout (bytes), single shared array assumed at LDS offset 0:
//   [0, 4096)          A tile: 4 row-tiles x (32 lanes x 16 bf16)
//   [4096, 12288)      B buffer 0: 8 col-tiles x 1KB (TDM destination)
//   [12288, 20480)     B buffer 1
#define LDS_A_BYTES   4096u
#define LDS_B_OFF     4096u
#define LDS_B_BYTES   8192u
#define LDS_TOTAL_US  ((LDS_A_BYTES + 2u * LDS_B_BYTES) / 2u)   // ushort count

__device__ __forceinline__ unsigned short f2bf(float f) {
    unsigned int u = __float_as_uint(f);
    u += 0x7FFFu + ((u >> 16) & 1u);      // round-to-nearest-even
    return (unsigned short)(u >> 16);
}

// pack two floats into one dword of bf16 (RNE); prefer v_cvt_pk_bf16_f32 when available
__device__ __forceinline__ unsigned int pack2bf(float a, float b) {
#if __has_builtin(__builtin_amdgcn_cvt_pk_bf16_f32)
    auto pk = __builtin_amdgcn_cvt_pk_bf16_f32(a, b);
    return __builtin_bit_cast(unsigned int, pk);
#else
    return (unsigned int)f2bf(a) | ((unsigned int)f2bf(b) << 16);
#endif
}

// ---- TDM: async-load one K-step of pre-swizzled B (8 rows x 1KB, stride 16KB) into LDS ----
// D# per CDNA5 ISA. 2D tile, data_size=4B, no pad/iterate/gather,
// workgroup_mask=0 (not in a cluster), VADDR2/VADDR3 omitted (<=2D tensor form).
__device__ __forceinline__ void tdm_load_b(const unsigned short* Wsw, int row0, unsigned lds_byte) {
    unsigned long long ga = (unsigned long long)(uintptr_t)Wsw + (unsigned long long)row0 * 1024ull;
    u32x4 g0;
    g0[0] = 1u;                                  // count=1, is_restore=0, gather off
    g0[1] = lds_byte;                            // lds_addr (bytes)
    g0[2] = (unsigned)(ga & 0xFFFFFFFFu);        // global_addr[31:0]
    g0[3] = (unsigned)(ga >> 32) | (2u << 30);   // global_addr[56:32] | type=2 ("image")
    u32x8 g1;
    g1[0] = (2u << 16);                          // workgroup_mask=0, data_size=2 (4 bytes)
    g1[1] = 0u;                                  // atomic barrier off; tensor_dim0[15:0] (of 2^30) = 0
    g1[2] = 0x4000u;                             // tensor_dim0[31:16]=0x4000; tensor_dim1[15:0]=0
    g1[3] = 0x4000u | (256u << 16);              // tensor_dim1[31:16]=0x4000; tile_dim0=256 dwords
    g1[4] = 8u;                                  // tile_dim1=8 rows; tile_dim2=0
    g1[5] = 4096u;                               // tensor_dim0_stride = 4096 dwords (16KB)
    g1[6] = 0u;
    g1[7] = 0u;
    asm volatile("tensor_load_to_lds %0, %1" :: "s"(g0), "s"(g1) : "memory");
}

// ---------------- copy h -> g (vectorized) ----------------
__global__ void gin_copy_kernel(const float* __restrict__ src, float* __restrict__ dst, int n4) {
    int i = blockIdx.x * blockDim.x + threadIdx.x;
    if (i < n4) reinterpret_cast<float4*>(dst)[i] = reinterpret_cast<const float4*>(src)[i];
}

__global__ void gin_zero_kernel(float* __restrict__ p, int n) {
    int i = blockIdx.x * blockDim.x + threadIdx.x;
    if (i < n) p[i] = 0.f;
}

// ---------------- edge scatter: g[dst] += h[src] ----------------
__global__ void gin_edge_scatter_kernel(const float* __restrict__ h, float* __restrict__ g,
                                        const int* __restrict__ ei, int E) {
    int gid = blockIdx.x * blockDim.x + threadIdx.x;      // E * 128 threads
    if (gid >= E * (HDIM / 4)) return;
    int e = gid >> 7;                 // HDIM/4 == 128
    int c = gid & 127;
    int s = ei[e];
    int d = ei[E + e];
    float4 v = reinterpret_cast<const float4*>(h + (size_t)s * HDIM)[c];
    float* out = g + (size_t)d * HDIM + c * 4;
    atomicAdd(out + 0, v.x);
    atomicAdd(out + 1, v.y);
    atomicAdd(out + 2, v.z);
    atomicAdd(out + 3, v.w);
}

// ---------------- weight pre-swizzle into WMMA B-fragment order (bf16) ----------------
// B-frag (this code's k order, matches A): lane L -> col = L%16, K-half = (L<16 ? 0 : 16); 16 bf16 per
// lane contiguous: out[((ct*16 + s)*32 + lane)*16 + j].
__global__ void gin_wswz_kernel(const float* __restrict__ w1, const float* __restrict__ w2,
                                unsigned short* __restrict__ o1, unsigned short* __restrict__ o2) {
    int idx = blockIdx.x * blockDim.x + threadIdx.x;      // 6 * 512 * 512
    if (idx >= 6 * HDIM * HDIM) return;
    int m = idx >> 18;                   // which matrix (0..5), HDIM*HDIM = 2^18
    int r = idx & (HDIM * HDIM - 1);
    int j    = r & 15;
    int lane = (r >> 4) & 31;
    int s    = (r >> 9) & 15;
    int ct   = r >> 13;
    int kb   = (lane < 16) ? 0 : 16;
    int k    = s * 32 + kb + j;
    int col  = ct * 16 + (lane & 15);
    const float* W = (m < 3) ? (w1 + (size_t)m * HDIM * HDIM)
                             : (w2 + (size_t)(m - 3) * HDIM * HDIM);
    unsigned short* O = (m < 3) ? (o1 + (size_t)m * HDIM * HDIM)
                                : (o2 + (size_t)(m - 3) * HDIM * HDIM);
    O[r] = f2bf(W[(size_t)k * HDIM + col]);
}

// ---------------- tiled bf16 WMMA GEMM, TDM-fed B, fused epilogue ----------------
// Block: 256 threads (8 waves). Block tile: 64 rows x 128 cols. Wave w owns col tile
// blockIdx.y*8 + w. B arrives via tensor_load_to_lds (double-buffered, prefetch s+1).
// MODE 1: y = BN( PReLU(acc + b) )    MODE 2: y = ReLU( PReLU(acc + b) )
template<int MODE>
__global__ __launch_bounds__(256) void gin_gemm_kernel(
        const float* __restrict__ X, const unsigned short* __restrict__ Wsw,
        const float* __restrict__ bias, const float* __restrict__ slope_p,
        const float* __restrict__ bn_gamma, const float* __restrict__ bn_beta,
        const float* __restrict__ bn_mean, const float* __restrict__ bn_var,
        float* __restrict__ Y, int nrows) {
    __shared__ __align__(32) unsigned short sLDS[LDS_TOTAL_US];

    const int t    = threadIdx.x;
    const int lane = t & 31;
    const int wave = t >> 5;
    const int rowbase = blockIdx.x * 64;
    const int ct      = blockIdx.y * 8 + wave;   // column tile (16 cols)
    const int brow0   = blockIdx.y * 128;        // first swizzled B row for this block

    // cooperative A-fill: each thread moves 8 contiguous floats -> 8 bf16 (16B LDS store)
    // A-frag (this code's k order, matches B): lane L -> row = L%16, K-base = (L<16 ? 0 : 16);
    // j 0..15 -> k = kb + j
    const int f_rt   = t >> 6;
    const int f_lane = (t >> 1) & 31;
    const int f_j0   = (t & 1) << 3;
    int f_row = rowbase + f_rt * 16 + (f_lane & 15);
    if (f_row > nrows - 1) f_row = nrows - 1;    // clamp (stores are guarded)
    const int f_k = ((f_lane < 16) ? 0 : 16) + f_j0;
    const float* __restrict__ Xrow = X + (size_t)f_row * HDIM + f_k;
    unsigned short* sDst = &sLDS[f_rt * 512 + f_lane * 16 + f_j0];

    const v8f vzero = {0.f, 0.f, 0.f, 0.f, 0.f, 0.f, 0.f, 0.f};
    v8f acc[4];
    #pragma unroll
    for (int i = 0; i < 4; ++i) acc[i] = vzero;

    // prologue: kick off TDM for K-step 0 into B buffer 0
    if (wave == 0) tdm_load_b(Wsw, brow0, LDS_B_OFF);

    for (int s = 0; s < KSTEPS; ++s) {
        const int buf = s & 1;
        const float4* xp = reinterpret_cast<const float4*>(Xrow + s * 32);
        float4 x0 = xp[0];
        float4 x1 = xp[1];
        __syncthreads();   // all waves finished reading sA and B buffer (buf^1) of iter s-1
        uint4 pk;
        pk.x = pack2bf(x0.x, x0.y);
        pk.y = pack2bf(x0.z, x0.w);
        pk.z = pack2bf(x1.x, x1.y);
        pk.w = pack2bf(x1.z, x1.w);
        *reinterpret_cast<uint4*>(sDst) = pk;
        if (wave == 0) {
            if (s + 1 < KSTEPS) {
                tdm_load_b(Wsw, brow0 + s + 1, LDS_B_OFF + (unsigned)((buf ^ 1) * LDS_B_BYTES));
                __builtin_amdgcn_s_wait_tensorcnt(1);   // TDM(s) complete, TDM(s+1) in flight
            } else {
                __builtin_amdgcn_s_wait_tensorcnt(0);
            }
        }
        __syncthreads();   // publish sA and B buffer buf

        const unsigned short* bp =
            &sLDS[(LDS_B_OFF / 2) + buf * (LDS_B_BYTES / 2) + wave * 512 + lane * 16];
        v16bf bfrag = *reinterpret_cast<const v16bf*>(bp);
        v16bf af[4];
        #pragma unroll
        for (int rt = 0; rt < 4; ++rt)
            af[rt] = *reinterpret_cast<const v16bf*>(&sLDS[rt * 512 + lane * 16]);
        #pragma unroll
        for (int rt = 0; rt < 4; ++rt)
            acc[rt] = __builtin_amdgcn_wmma_f32_16x16x32_bf16(
                false, af[rt], false, bfrag, (short)0, acc[rt], false, false);
    }

    // epilogue — C/D layout: VGPR r -> M = r (lanes 0-15) or r+8 (lanes 16-31); col = lane%16
    const int ncol  = ct * 16 + (lane & 15);
    const float b   = bias[ncol];
    const float a   = slope_p[0];
    float scale = 1.f, shift = 0.f;
    if (MODE == 1) {
        scale = bn_gamma[ncol] * rsqrtf(bn_var[ncol] + 1e-5f);
        shift = bn_beta[ncol] - bn_mean[ncol] * scale;
    }
    const int rbase = rowbase + ((lane < 16) ? 0 : 8);
    #pragma unroll
    for (int rt = 0; rt < 4; ++rt) {
        #pragma unroll
        for (int r = 0; r < 8; ++r) {
            int row = rbase + rt * 16 + r;
            if (row < nrows) {
                float v = acc[rt][r] + b;
                v = (v >= 0.f) ? v : a * v;           // PReLU
                if (MODE == 1) v = v * scale + shift; // folded BatchNorm (eval)
                else           v = fmaxf(v, 0.f);     // ReLU after PReLU
                Y[(size_t)row * HDIM + ncol] = v;
            }
        }
    }
}

// ---------------- global add pool: pooled[batch[i]] += h[i] ----------------
__global__ void gin_pool_kernel(const float* __restrict__ h, const int* __restrict__ batch,
                                float* __restrict__ pooled, int N) {
    int gid = blockIdx.x * blockDim.x + threadIdx.x;     // N * 128
    if (gid >= N * (HDIM / 4)) return;
    int i = gid >> 7;
    int c = gid & 127;
    float4 v = reinterpret_cast<const float4*>(h + (size_t)i * HDIM)[c];
    float* out = pooled + (size_t)batch[i] * HDIM + c * 4;
    atomicAdd(out + 0, v.x);
    atomicAdd(out + 1, v.y);
    atomicAdd(out + 2, v.z);
    atomicAdd(out + 3, v.w);
}

// ---------------- fc1: relu([pooled | stats | adj] @ W + b) ----------------
__global__ void gin_fc1_kernel(const float* __restrict__ pooled, const float* __restrict__ stats,
                               const float* __restrict__ adj, const float* __restrict__ w,
                               const float* __restrict__ b, float* __restrict__ xl,
                               int ncond, int nadj) {
    const int g = blockIdx.x;
    const int n = threadIdx.x;       // 256 = H/2 outputs
    float acc = b[n];
    const float* pg = pooled + (size_t)g * HDIM;
    for (int f = 0; f < HDIM; ++f)  acc += pg[f] * w[(size_t)f * 256 + n];
    const float* sg = stats + (size_t)g * ncond;
    for (int f = 0; f < ncond; ++f) acc += sg[f] * w[(size_t)(HDIM + f) * 256 + n];
    const float* ag = adj + (size_t)g * nadj;
    for (int f = 0; f < nadj; ++f)  acc += ag[f] * w[(size_t)(HDIM + ncond + f) * 256 + n];
    xl[(size_t)g * 256 + n] = fmaxf(acc, 0.f);
}

// ---------------- fc2: sigmoid(x_l @ w + b) ----------------
__global__ void gin_fc2_kernel(const float* __restrict__ xl, const float* __restrict__ w,
                               const float* __restrict__ b, float* __restrict__ out, int G) {
    int g = blockIdx.x * blockDim.x + threadIdx.x;
    if (g >= G) return;
    float acc = b[0];
    const float* xg = xl + (size_t)g * 256;
    for (int n = 0; n < 256; ++n) acc += xg[n] * w[n];
    out[g] = 1.f / (1.f + expf(-acc));
}

extern "C" void kernel_launch(void* const* d_in, const int* in_sizes, int n_in,
                              void* d_out, int out_size, void* d_ws, size_t ws_size,
                              hipStream_t stream) {
    const float* x     = (const float*)d_in[0];
    const int*   ei    = (const int*)  d_in[1];
    const int*   batch = (const int*)  d_in[2];
    const float* stats = (const float*)d_in[3];
    const float* adj   = (const float*)d_in[4];
    const float* w1    = (const float*)d_in[5];
    const float* b1    = (const float*)d_in[6];
    const float* a1    = (const float*)d_in[7];
    const float* bng   = (const float*)d_in[8];
    const float* bnb   = (const float*)d_in[9];
    const float* bnm   = (const float*)d_in[10];
    const float* bnv   = (const float*)d_in[11];
    const float* w2    = (const float*)d_in[12];
    const float* b2    = (const float*)d_in[13];
    const float* a2    = (const float*)d_in[14];
    const float* fc1w  = (const float*)d_in[15];
    const float* fc1b  = (const float*)d_in[16];
    const float* fc2w  = (const float*)d_in[17];
    const float* fc2b  = (const float*)d_in[18];
    float* out = (float*)d_out;

    const int N     = in_sizes[0] / HDIM;   // 20000
    const int E     = in_sizes[1] / 2;      // 320000
    const int G     = in_sizes[3] / 8;      // 400
    const int NCOND = 8;
    const int NADJ  = in_sizes[4] / G;      // 2500

    // workspace layout (~86 MB)
    float*          bufA   = (float*)d_ws;
    float*          bufB   = bufA + (size_t)N * HDIM;
    unsigned short* w1sw   = (unsigned short*)(bufB + (size_t)N * HDIM);
    unsigned short* w2sw   = w1sw + (size_t)NLAYER * HDIM * HDIM;
    float*          pooled = (float*)(w2sw + (size_t)NLAYER * HDIM * HDIM);

    // 1) pre-swizzle all six weight matrices to bf16 fragment layout
    {
        int total = 6 * HDIM * HDIM;
        gin_wswz_kernel<<<(total + 255) / 256, 256, 0, stream>>>(w1, w2, w1sw, w2sw);
    }

    // 2) GIN layers with ping-pong buffers
    const float* h = x;
    for (int l = 0; l < NLAYER; ++l) {
        float* g = (h == (const float*)bufA) ? bufB : bufA;
        int n4 = N * HDIM / 4;
        gin_copy_kernel<<<(n4 + 255) / 256, 256, 0, stream>>>(h, g, n4);
        int etot = E * (HDIM / 4);
        gin_edge_scatter_kernel<<<(etot + 255) / 256, 256, 0, stream>>>(h, g, ei, E);

        float* tbuf = (g == bufA) ? bufB : bufA;
        dim3 grid((N + 63) / 64, HDIM / 128);
        gin_gemm_kernel<1><<<grid, 256, 0, stream>>>(
            g, w1sw + (size_t)l * HDIM * HDIM, b1 + l * HDIM, a1 + l,
            bng + l * HDIM, bnb + l * HDIM, bnm + l * HDIM, bnv + l * HDIM, tbuf, N);
        gin_gemm_kernel<2><<<grid, 256, 0, stream>>>(
            tbuf, w2sw + (size_t)l * HDIM * HDIM, b2 + l * HDIM, a2 + l,
            nullptr, nullptr, nullptr, nullptr, g, N);
        h = g;
    }

    // 3) global add pool
    gin_zero_kernel<<<(G * HDIM + 255) / 256, 256, 0, stream>>>(pooled, G * HDIM);
    int ptot = N * (HDIM / 4);
    gin_pool_kernel<<<(ptot + 255) / 256, 256, 0, stream>>>(h, batch, pooled, N);

    // 4) head: d_out = [out(G) | x_l(G x 256)], tuple order
    float* xl_out = out + G;
    gin_fc1_kernel<<<G, 256, 0, stream>>>(pooled, stats, adj, fc1w, fc1b, xl_out, NCOND, NADJ);
    gin_fc2_kernel<<<(G + 255) / 256, 256, 0, stream>>>(xl_out, fc2w, fc2b, out, G);
}
